// circle_fine_grained_extractor_47201690583535
// MI455X (gfx1250) — matrix operand order repaired and checked by reading, not compiled, not run on hardware
//
#include <hip/hip_runtime.h>
#include <hip/hip_bf16.h>

// ---------------- problem dims ----------------
#define BN 32       // batch
#define QN 16       // queries
#define CN 2048     // channels
#define PN 288      // tokens per patch (H*W)
#define NI 4        // num instance
#define N3 (NI*PN)  // 1152 tokens for f_cor
#define SCALE_F 0.022097086912079608f  // 2048^-0.5

typedef __bf16 bf16_t;
typedef __attribute__((ext_vector_type(16))) __bf16 v16bf;
typedef __attribute__((ext_vector_type(8)))  float  v8f;
typedef __attribute__((ext_vector_type(4)))  int    v4i;

#define DEV static __device__ __forceinline__

// ---------------- async global->LDS staging (gfx1250) ----------------
#if __has_builtin(__builtin_amdgcn_global_load_async_to_lds_b128) && \
    __has_builtin(__builtin_amdgcn_s_wait_asynccnt)
#define ASYNC_LDS 1
typedef __attribute__((address_space(1))) v4i g_v4i;  // global int4
typedef __attribute__((address_space(3))) v4i l_v4i;  // LDS int4
DEV void async_cp16(const float* g, float* l) {
  __builtin_amdgcn_global_load_async_to_lds_b128((g_v4i*)g, (l_v4i*)l, 0, 0);
}
DEV void async_wait_stage(int more_in_flight) {
  if (more_in_flight) __builtin_amdgcn_s_wait_asynccnt(4);
  else                __builtin_amdgcn_s_wait_asynccnt(0);
}
#else
#define ASYNC_LDS 0
DEV void async_cp16(const float* g, float* l) { *(float4*)l = *(const float4*)g; }
DEV void async_wait_stage(int) {}
#endif

// ---------------- WMMA helpers (CDNA5, wave32) ----------------
DEV v8f wmma_bf16(v16bf a, v16bf b, v8f c) {
  return __builtin_amdgcn_wmma_f32_16x16x32_bf16(false, a, false, b, (short)0, c,
                                                 false, false);
}

// A fragment: 16x32, lane(0..15)=row M, K = e + 16*(lane>=16)
DEV v16bf load_a_f32(const float* base, long sm, long sk, int k0) {
  int lane = threadIdx.x & 31;
  int m = lane & 15;
  int koff = (lane >> 4) * 16;
  const float* p = base + (long)m * sm + (long)(k0 + koff) * sk;
  v16bf a;
#pragma unroll
  for (int e = 0; e < 8; ++e) a[e] = (__bf16)p[(long)e * sk];
#pragma unroll
  for (int e = 0; e < 8; ++e) a[8 + e] = (__bf16)p[(long)(8 + e) * sk];
  return a;
}
DEV v16bf load_a_bf(const bf16_t* base, long sm, long sk, int k0) {
  int lane = threadIdx.x & 31;
  int m = lane & 15;
  int koff = (lane >> 4) * 16;
  const bf16_t* p = base + (long)m * sm + (long)(k0 + koff) * sk;
  v16bf a;
#pragma unroll
  for (int e = 0; e < 8; ++e) a[e] = p[(long)e * sk];
#pragma unroll
  for (int e = 0; e < 8; ++e) a[8 + e] = p[(long)(8 + e) * sk];
  return a;
}
// B fragment: 32x16, lane(0..15)=col N, K = e + 16*(lane>=16)
DEV v16bf load_b_f32(const float* base, long sn, long sk, int k0) {
  int lane = threadIdx.x & 31;
  int n = lane & 15;
  int koff = (lane >> 4) * 16;
  const float* p = base + (long)n * sn + (long)(k0 + koff) * sk;
  v16bf b;
#pragma unroll
  for (int e = 0; e < 16; ++e) b[e] = (__bf16)p[(long)e * sk];
  return b;
}
DEV v16bf load_b_bf(const bf16_t* base, long sn, long sk, int k0) {
  int lane = threadIdx.x & 31;
  int n = lane & 15;
  int koff = (lane >> 4) * 16;
  const bf16_t* p = base + (long)n * sn + (long)(k0 + koff) * sk;
  v16bf b;
#pragma unroll
  for (int e = 0; e < 16; ++e) b[e] = p[(long)e * sk];
  return b;
}
// C/D: lane(0..15)=N, vgpr v -> M = v + 8*(lane>=16)
DEV void store_tile_f32(float* out, long ldc, const v8f& acc) {
  int lane = threadIdx.x & 31;
  int n = lane & 15, mh = (lane >> 4) * 8;
#pragma unroll
  for (int v = 0; v < 8; ++v) out[(long)(mh + v) * ldc + n] = acc[v];
}

// ---------------- elementwise / prep kernels ----------------

// transpose x[b,c,n] -> xs_bf[b,n,c] (bf16) + inverse L2 norm per token
__global__ void k_prep(const float* __restrict__ x, bf16_t* __restrict__ xs_bf,
                       float* __restrict__ rn) {
  int token = blockIdx.x;          // b*PN+n
  int b = token / PN, n = token % PN;
  const float* xp = x + (long)b * CN * PN + n;
  bf16_t* op = xs_bf + (long)token * CN;
  __shared__ float red[256];
  float ss = 0.f;
  for (int c = threadIdx.x; c < CN; c += blockDim.x) {
    float v = xp[(long)c * PN];
    op[c] = (__bf16)v;
    ss += v * v;
  }
  red[threadIdx.x] = ss; __syncthreads();
  for (int s = blockDim.x / 2; s > 0; s >>= 1) {
    if (threadIdx.x < s) red[threadIdx.x] += red[threadIdx.x + s];
    __syncthreads();
  }
  if (threadIdx.x == 0) rn[token] = 1.0f / fmaxf(sqrtf(red[0]), 1e-12f);
}

__global__ void k_cam(const int* __restrict__ ids, const float* __restrict__ table,
                      float* __restrict__ cam) {
  int b = blockIdx.x;
  const float* src = table + (long)(ids[b] - 1) * CN;
  float* dst = cam + (long)b * CN;
  for (int c = threadIdx.x; c < CN; c += blockDim.x) dst[c] = src[c];
}

// stable argsort by sub (False first) -> token-group map b3[k][j]
__global__ void k_perm(const unsigned char* __restrict__ sub, int* __restrict__ b3) {
  if (threadIdx.x == 0 && blockIdx.x == 0) {
    int perm[BN]; int idx = 0;
    for (int b = 0; b < BN; ++b) if (!sub[b]) perm[idx++] = b;
    for (int b = 0; b < BN; ++b) if (sub[b])  perm[idx++] = b;
    for (int k = 0; k < BN; ++k)
      for (int j = 0; j < NI; ++j) b3[k * NI + j] = (perm[k] / NI) * NI + j;
  }
}

__global__ void k_wsm(const float* __restrict__ w, float* __restrict__ wsm) {
  if (threadIdx.x == 0 && blockIdx.x == 0) {
    float m = w[0];
    for (int i = 1; i < QN; ++i) m = fmaxf(m, w[i]);
    float e[QN], s = 0.f;
    for (int i = 0; i < QN; ++i) { e[i] = __expf(w[i] - m); s += e[i]; }
    for (int i = 0; i < QN; ++i) wsm[i] = e[i] / s;
  }
}

// generic in-place row softmax
__global__ void k_softmax(float* __restrict__ data, int L) {
  float* p = data + (long)blockIdx.x * L;
  __shared__ float red[256];
  float m = -3.4e38f;
  for (int i = threadIdx.x; i < L; i += blockDim.x) m = fmaxf(m, p[i]);
  red[threadIdx.x] = m; __syncthreads();
  for (int s = blockDim.x / 2; s > 0; s >>= 1) {
    if (threadIdx.x < s) red[threadIdx.x] = fmaxf(red[threadIdx.x], red[threadIdx.x + s]);
    __syncthreads();
  }
  m = red[0]; __syncthreads();
  float sum = 0.f;
  for (int i = threadIdx.x; i < L; i += blockDim.x) {
    float e = __expf(p[i] - m); p[i] = e; sum += e;
  }
  red[threadIdx.x] = sum; __syncthreads();
  for (int s = blockDim.x / 2; s > 0; s >>= 1) {
    if (threadIdx.x < s) red[threadIdx.x] += red[threadIdx.x + s];
    __syncthreads();
  }
  float inv = 1.0f / red[0];
  for (int i = threadIdx.x; i < L; i += blockDim.x) p[i] *= inv;
}

// rows of 2048: LayerNorm (no affine)
__global__ void k_ln(const float* __restrict__ in, float* __restrict__ out) {
  const float* p = in + (long)blockIdx.x * CN;
  float* o = out + (long)blockIdx.x * CN;
  __shared__ float r1[256], r2[256];
  float s = 0.f, s2 = 0.f;
  for (int i = threadIdx.x; i < CN; i += blockDim.x) { float v = p[i]; s += v; s2 += v * v; }
  r1[threadIdx.x] = s; r2[threadIdx.x] = s2; __syncthreads();
  for (int st = blockDim.x / 2; st > 0; st >>= 1) {
    if (threadIdx.x < st) { r1[threadIdx.x] += r1[threadIdx.x + st]; r2[threadIdx.x] += r2[threadIdx.x + st]; }
    __syncthreads();
  }
  float mean = r1[0] * (1.0f / CN);
  float var = r2[0] * (1.0f / CN) - mean * mean;
  float rstd = rsqrtf(var + 1e-5f);
  for (int i = threadIdx.x; i < CN; i += blockDim.x) o[i] = (p[i] - mean) * rstd;
}

__global__ void k_meanx(const float* __restrict__ f_rel, float* __restrict__ meanx) {
  int b = blockIdx.y;
  int c = blockIdx.x * blockDim.x + threadIdx.x;
  float s = 0.f;
  for (int q = 0; q < QN; ++q) s += f_rel[((long)b * QN + q) * CN + c];
  meanx[(long)b * CN + c] = s * (1.0f / QN);
}

// per (b,n): LN stats of proto[n,:]-meanx[b,:]
__global__ void k_dstats(const float* __restrict__ proto, const float* __restrict__ meanx,
                         float* __restrict__ mu, float* __restrict__ rstd) {
  int n = blockIdx.x, b = blockIdx.y;
  const float* pp = proto + (long)n * CN;
  const float* mx = meanx + (long)b * CN;
  __shared__ float r1[256], r2[256];
  float s = 0.f, s2 = 0.f;
  for (int c = threadIdx.x; c < CN; c += blockDim.x) {
    float y = pp[c] - mx[c]; s += y; s2 += y * y;
  }
  r1[threadIdx.x] = s; r2[threadIdx.x] = s2; __syncthreads();
  for (int st = blockDim.x / 2; st > 0; st >>= 1) {
    if (threadIdx.x < st) { r1[threadIdx.x] += r1[threadIdx.x + st]; r2[threadIdx.x] += r2[threadIdx.x + st]; }
    __syncthreads();
  }
  if (threadIdx.x == 0) {
    float mean = r1[0] * (1.0f / CN);
    float var = r2[0] * (1.0f / CN) - mean * mean;
    mu[(long)b * CN + n] = mean;
    rstd[(long)b * CN + n] = rsqrtf(var + 1e-5f);
  }
}

// per row r=b*16+q: sumt = sum(t), tmx = dot(t, meanx[b])
__global__ void k_taux(const float* __restrict__ t, const float* __restrict__ meanx,
                       float* __restrict__ sumt, float* __restrict__ tmx) {
  int r = blockIdx.x, b = r >> 4;
  const float* tp = t + (long)r * CN;
  const float* mx = meanx + (long)b * CN;
  __shared__ float r1[256], r2[256];
  float s1 = 0.f, s2 = 0.f;
  for (int c = threadIdx.x; c < CN; c += blockDim.x) { float v = tp[c]; s1 += v; s2 += v * mx[c]; }
  r1[threadIdx.x] = s1; r2[threadIdx.x] = s2; __syncthreads();
  for (int st = blockDim.x / 2; st > 0; st >>= 1) {
    if (threadIdx.x < st) { r1[threadIdx.x] += r1[threadIdx.x + st]; r2[threadIdx.x] += r2[threadIdx.x + st]; }
    __syncthreads();
  }
  if (threadIdx.x == 0) { sumt[r] = r1[0]; tmx[r] = r2[0]; }
}

// per row: w = attd * rstd[b,n]; S1 = sum(w); S2 = sum(w*mu)
__global__ void k_wmul(float* __restrict__ attd, const float* __restrict__ rstd,
                       const float* __restrict__ mu, float* __restrict__ S1,
                       float* __restrict__ S2) {
  int r = blockIdx.x, b = r >> 4;
  float* p = attd + (long)r * CN;
  const float* rs = rstd + (long)b * CN;
  const float* mup = mu + (long)b * CN;
  __shared__ float r1[256], r2[256];
  float s1 = 0.f, s2 = 0.f;
  for (int n = threadIdx.x; n < CN; n += blockDim.x) {
    float w = p[n] * rs[n]; p[n] = w; s1 += w; s2 += w * mup[n];
  }
  r1[threadIdx.x] = s1; r2[threadIdx.x] = s2; __syncthreads();
  for (int st = blockDim.x / 2; st > 0; st >>= 1) {
    if (threadIdx.x < st) { r1[threadIdx.x] += r1[threadIdx.x + st]; r2[threadIdx.x] += r2[threadIdx.x + st]; }
    __syncthreads();
  }
  if (threadIdx.x == 0) { S1[r] = r1[0]; S2[r] = r2[0]; }
}

// camterm[item, j, q] = dot(u[item or shared][q], cam[b3[item][j] or item])
__global__ void k_camdot(const float* __restrict__ u, long uStride,
                         const float* __restrict__ cam, const int* __restrict__ b3,
                         int ngroups, float* __restrict__ out) {
  int item = blockIdx.x / (ngroups * QN);
  int rem = blockIdx.x % (ngroups * QN);
  int j = rem / QN, q = rem % QN;
  int g = b3 ? b3[item * NI + j] : item;
  const float* up = u + (long)item * uStride + (long)q * CN;
  const float* cp = cam + (long)g * CN;
  __shared__ float red[256];
  float s = 0.f;
  for (int c = threadIdx.x; c < CN; c += blockDim.x) s += up[c] * cp[c];
  red[threadIdx.x] = s; __syncthreads();
  for (int st = blockDim.x / 2; st > 0; st >>= 1) {
    if (threadIdx.x < st) red[threadIdx.x] += red[threadIdx.x + st];
    __syncthreads();
  }
  if (threadIdx.x == 0) out[blockIdx.x] = red[0];
}

// ---------------- WMMA GEMM kernels ----------------

// cosine similarity Gram: S[b,p,q] = rn[p]*rn[q] * <xs[p], xs[q]>
__global__ void k_gram(const bf16_t* __restrict__ xs_bf, const float* __restrict__ rn,
                       float* __restrict__ S) {
  int b = blockIdx.y;
  int tp = blockIdx.x / (PN / 16), tq = blockIdx.x % (PN / 16);
  const bf16_t* base = xs_bf + (long)b * PN * CN;
  const bf16_t* pA = base + (long)tp * 16 * CN;
  const bf16_t* pB = base + (long)tq * 16 * CN;
  v8f acc = {};
  for (int k0 = 0; k0 < CN; k0 += 32) {
    __builtin_prefetch(pA + (long)(k0 + 64), 0, 1);
    v16bf a = load_a_bf(pA, CN, 1, k0);
    v16bf bb = load_b_bf(pB, CN, 1, k0);
    acc = wmma_bf16(a, bb, acc);
  }
  int lane = threadIdx.x & 31;
  int n = lane & 15, mh = (lane >> 4) * 8;
  const float* rnb = rn + (long)b * PN;
  float rq = rnb[tq * 16 + n];
  float* out = S + (long)b * PN * PN + (long)(tp * 16 + mh) * PN + tq * 16 + n;
#pragma unroll
  for (int v = 0; v < 8; ++v) out[(long)v * PN] = acc[v] * rnb[tp * 16 + mh + v] * rq;
}

// g2 = R @ xs : [PN,PN] x [PN,CN], output bf16
__global__ void k_g2(const float* __restrict__ R, const bf16_t* __restrict__ xs_bf,
                     bf16_t* __restrict__ g2_bf) {
  int b = blockIdx.y;
  int tp = blockIdx.x / (CN / 16), tc = blockIdx.x % (CN / 16);
  const float* pA = R + (long)b * PN * PN + (long)tp * 16 * PN;
  const bf16_t* pB = xs_bf + (long)b * PN * CN + tc * 16;  // B[k=n', c]
  v8f acc = {};
  for (int k0 = 0; k0 < PN; k0 += 32) {
    v16bf a = load_a_f32(pA, PN, 1, k0);
    v16bf bb = load_b_bf(pB, 1, CN, k0);
    acc = wmma_bf16(a, bb, acc);
  }
  int lane = threadIdx.x & 31;
  int n = lane & 15, mh = (lane >> 4) * 8;
  bf16_t* out = g2_bf + (long)b * PN * CN + (long)(tp * 16 + mh) * CN + tc * 16 + n;
#pragma unroll
  for (int v = 0; v < 8; ++v) out[(long)v * CN] = (__bf16)acc[v];
}

// WcombT[j,i] = sum_d Wa[d,i]*Wb[d,j]   (i.e. Wcomb = Wa^T @ Wb, stored transposed)
// LDS-staged, double-buffered, async global->LDS copies; 64x64 tile per 8-wave block.
__global__ void __launch_bounds__(256) k_wcomb_lds(const float* __restrict__ Wa,
                                                   const float* __restrict__ Wb,
                                                   bf16_t* __restrict__ outT) {
  constexpr int PITCH = 68;  // 64 + pad, keeps 16B alignment, avoids bank conflicts
  __shared__ __align__(16) float As[2][32][PITCH];
  __shared__ __align__(16) float Bs[2][32][PITCH];
  int i0 = blockIdx.y * 64, j0 = blockIdx.x * 64;
  int tid = threadIdx.x;
  int wave = tid >> 5, lane = tid & 31;
  int si = wave >> 1, sj = (wave & 1) * 2;   // wave -> 16-row strip x two 16-col strips

  auto issue = [&](int buf, int k0) {
    // 32 K-rows x 64 floats per matrix = 512 16B chunks; 2 per thread per matrix
    for (int c = tid; c < 512; c += 256) {
      int d = c >> 4, q = (c & 15) * 4;
      async_cp16(Wa + (long)(k0 + d) * CN + i0 + q, &As[buf][d][q]);
      async_cp16(Wb + (long)(k0 + d) * CN + j0 + q, &Bs[buf][d][q]);
    }
  };

  issue(0, 0);
  v8f acc0 = {}, acc1 = {};
  const int steps = CN / 32;
  for (int s = 0; s < steps; ++s) {
    int buf = s & 1;
    __syncthreads();                         // prior compute done before refill
    if (s + 1 < steps) issue(buf ^ 1, (s + 1) * 32);
    async_wait_stage(s + 1 < steps);         // stage s complete (<=4 newer in flight)
    __syncthreads();                         // LDS visible to all waves

    int m = lane & 15;
    int koff8 = (lane >> 4) * 8;
    int koff16 = (lane >> 4) * 16;
    v16bf a, b0, b1;
#pragma unroll
    for (int e = 0; e < 8; ++e) a[e] = (__bf16)As[buf][koff16 + e][si * 16 + m];
#pragma unroll
    for (int e = 0; e < 8; ++e) a[8 + e] = (__bf16)As[buf][koff16 + 8 + e][si * 16 + m];
#pragma unroll
    for (int e = 0; e < 16; ++e) b0[e] = (__bf16)Bs[buf][koff16 + e][sj * 16 + m];
#pragma unroll
    for (int e = 0; e < 16; ++e) b1[e] = (__bf16)Bs[buf][koff16 + e][(sj + 1) * 16 + m];
    acc0 = wmma_bf16(a, b0, acc0);
    acc1 = wmma_bf16(a, b1, acc1);
  }
  int n = lane & 15, mh = (lane >> 4) * 8;
#pragma unroll
  for (int v = 0; v < 8; ++v) {
    int i = i0 + si * 16 + mh + v;
    outT[(long)(j0 + sj * 16 + n) * CN + i] = (__bf16)acc0[v];
    outT[(long)(j0 + (sj + 1) * 16 + n) * CN + i] = (__bf16)acc1[v];
  }
}

// Y[item] = X[item] @ W  (M=16, K=CN, N=CN); W given TRANSPOSED: WT[n,k] (bf16)
__global__ void k_rowgemm(const float* __restrict__ X, long xItemStride,
                          const bf16_t* __restrict__ WT, float* __restrict__ Y) {
  int item = blockIdx.y, tn = blockIdx.x;
  const float* pA = X + (long)item * xItemStride;
  const bf16_t* pB = WT + (long)tn * 16 * CN;  // B[k,n] = WT[n,k] -> contiguous in k
  v8f acc = {};
  for (int k0 = 0; k0 < CN; k0 += 32) {
    __builtin_prefetch(pB + (long)(k0 + 64), 0, 1);
    v16bf a = load_a_f32(pA, CN, 1, k0);
    v16bf bb = load_b_bf(pB, CN, 1, k0);
    acc = wmma_bf16(a, bb, acc);
  }
  store_tile_f32(Y + (long)item * QN * CN + tn * 16, CN, acc);
}

// logits: scores[item,q,(j,n)] = SCALE*( u[item][q] . g2[g(j)][n] + camterm[item,j,q] )
__global__ void k_scores16(const float* __restrict__ u, long uStride,
                           const bf16_t* __restrict__ g2_bf, const int* __restrict__ b3,
                           int ngroups, const float* __restrict__ camterm,
                           float* __restrict__ scores) {
  int item = blockIdx.y;
  int tile = blockIdx.x;               // ngroups * (PN/16)
  int j = tile / (PN / 16), tn = tile % (PN / 16);
  int g = b3 ? b3[item * NI + j] : item;
  const float* pA = u + (long)item * uStride;
  const bf16_t* pB = g2_bf + (long)g * PN * CN + (long)tn * 16 * CN;  // B[k=c, n] = g2[n,c]
  v8f acc = {};
  for (int k0 = 0; k0 < CN; k0 += 32) {
    v16bf a = load_a_f32(pA, CN, 1, k0);
    v16bf bb = load_b_bf(pB, CN, 1, k0);
    acc = wmma_bf16(a, bb, acc);
  }
  int Ntot = ngroups * PN;
  int lane = threadIdx.x & 31;
  int n = lane & 15, mh = (lane >> 4) * 8;
  const float* ct = camterm + (long)(item * ngroups + j) * QN;
  float* out = scores + (long)item * QN * Ntot + (long)j * PN + tn * 16 + n;
#pragma unroll
  for (int v = 0; v < 8; ++v)
    out[(long)(mh + v) * Ntot] = SCALE_F * (acc[v] + ct[mh + v]);
}

// F[item] = att @ g2(gathered)   (M=16, K=Ntot, N=CN)
__global__ void k_attout(const float* __restrict__ att, const bf16_t* __restrict__ g2_bf,
                         const int* __restrict__ b3, int ngroups, float* __restrict__ out) {
  int item = blockIdx.y, tc = blockIdx.x;
  int Ntot = ngroups * PN;
  const float* pA = att + (long)item * QN * Ntot;
  v8f acc = {};
  for (int k0 = 0; k0 < Ntot; k0 += 32) {
    int j = k0 / PN;                 // PN=288 divisible by 32 -> tile within one group
    int nn0 = k0 - j * PN;
    int g = b3 ? b3[item * NI + j] : item;
    const bf16_t* pB = g2_bf + (long)g * PN * CN + (long)nn0 * CN + tc * 16;  // B[k,c]
    v16bf a = load_a_f32(pA, Ntot, 1, k0);
    v16bf bb = load_b_bf(pB, 1, CN, 0);
    acc = wmma_bf16(a, bb, acc);
  }
  store_tile_f32(out + (long)item * QN * CN + tc * 16, CN, acc);
}

// deltaor logits: scd[b,q,n] = SCALE * rstd[b,n] * ( t[b,q].proto[n] - tmx[b,q] - mu[b,n]*sumt[b,q] )
__global__ void k_dscores(const float* __restrict__ t, const float* __restrict__ proto,
                          const float* __restrict__ mu, const float* __restrict__ rstd,
                          const float* __restrict__ sumt, const float* __restrict__ tmx,
                          float* __restrict__ scd) {
  int b = blockIdx.y, tn = blockIdx.x;
  const float* pA = t + (long)b * QN * CN;
  const float* pB = proto + (long)tn * 16 * CN;  // B[k=c, n] = proto[n,c]
  v8f acc = {};
  for (int k0 = 0; k0 < CN; k0 += 32) {
    __builtin_prefetch(pB + (long)(k0 + 64), 0, 1);
    v16bf a = load_a_f32(pA, CN, 1, k0);
    v16bf bb = load_b_f32(pB, CN, 1, k0);
    acc = wmma_bf16(a, bb, acc);
  }
  int lane = threadIdx.x & 31;
  int n = lane & 15, mh = (lane >> 4) * 8;
  int n_ = tn * 16 + n;
  float rs = rstd[(long)b * CN + n_];
  float muv = mu[(long)b * CN + n_];
#pragma unroll
  for (int v = 0; v < 8; ++v) {
    int q = mh + v;
    scd[((long)b * QN + q) * CN + n_] =
        SCALE_F * rs * (acc[v] - tmx[b * QN + q] - muv * sumt[b * QN + q]);
  }
}

// f_pro_raw = f_rel + w@proto - S1*meanx - S2 ; weighted copy to output
__global__ void k_outd(const float* __restrict__ w, const float* __restrict__ proto,
                       const float* __restrict__ f_rel, const float* __restrict__ meanx,
                       const float* __restrict__ S1, const float* __restrict__ S2,
                       const float* __restrict__ wsm, float* __restrict__ f_pro_raw,
                       float* __restrict__ out_fpro) {
  int b = blockIdx.y, tc = blockIdx.x;
  const float* pA = w + (long)b * QN * CN;       // [16, CN] over n
  const float* pB = proto + tc * 16;             // B[k=n, c] = proto[n,c]
  v8f acc = {};
  for (int k0 = 0; k0 < CN; k0 += 32) {
    __builtin_prefetch(pB + (long)(k0 + 32) * CN, 0, 1);
    v16bf a = load_a_f32(pA, CN, 1, k0);
    v16bf bb = load_b_f32(pB, 1, CN, k0);
    acc = wmma_bf16(a, bb, acc);
  }
  int lane = threadIdx.x & 31;
  int n = lane & 15, mh = (lane >> 4) * 8;
  int c_ = tc * 16 + n;
  float mx = meanx[(long)b * CN + c_];
#pragma unroll
  for (int v = 0; v < 8; ++v) {
    int q = mh + v;
    long idx = ((long)b * QN + q) * CN + c_;
    float val = f_rel[idx] + acc[v] - S1[b * QN + q] * mx - S2[b * QN + q];
    f_pro_raw[idx] = val;
    out_fpro[idx] = wsm[q] * val;
  }
}

// ---------------- host launch ----------------
extern "C" void kernel_launch(void* const* d_in, const int* in_sizes, int n_in,
                              void* d_out, int out_size, void* d_ws, size_t ws_size,
                              hipStream_t stream) {
  (void)in_sizes; (void)n_in; (void)out_size; (void)ws_size;
  const float* x        = (const float*)d_in[0];
  const int*   cam_ids  = (const int*)d_in[1];
  const unsigned char* sub = (const unsigned char*)d_in[2];  // jnp bool -> 1 byte
  const float* query_v  = (const float*)d_in[3];
  const float* proto    = (const float*)d_in[4];
  const float* weights  = (const float*)d_in[5];
  const float* cam_tab  = (const float*)d_in[6];
  // d_in[7] pos_embed unused: cam_embed is always present in this graph
  const float* Wcq = (const float*)d_in[8];
  const float* Wcg = (const float*)d_in[9];
  const float* Wdq = (const float*)d_in[10];
  const float* Wdg = (const float*)d_in[11];

  float* f_rel = (float*)d_out;                       // [32,16,2048]
  float* f_pro = f_rel + (long)BN * QN * CN;          // weighted
  float* f_rec = f_pro + (long)BN * QN * CN;
  float* f_cor = f_rec + (long)BN * QN * CN;

  // deterministic workspace carve-out
  char* wp = (char*)d_ws;
  auto alloc = [&](size_t bytes) -> void* {
    void* p = (void*)wp;
    wp += (bytes + 255) & ~(size_t)255;
    return p;
  };
  bf16_t* xs_bf   = (bf16_t*)alloc((size_t)BN * PN * CN * 2);
  bf16_t* g2_bf   = (bf16_t*)alloc((size_t)BN * PN * CN * 2);
  bf16_t* WcombT  = (bf16_t*)alloc((size_t)CN * CN * 2);
  bf16_t* WdcombT = (bf16_t*)alloc((size_t)CN * CN * 2);
  float*  S       = (float*)alloc((size_t)BN * PN * PN * 4);
  float*  rn      = (float*)alloc((size_t)BN * PN * 4);
  float*  cam     = (float*)alloc((size_t)BN * CN * 4);
  float*  u1      = (float*)alloc((size_t)QN * CN * 4);
  float*  u2      = (float*)alloc((size_t)BN * QN * CN * 4);
  float*  qln     = (float*)alloc((size_t)BN * QN * CN * 4);
  float*  tmat    = (float*)alloc((size_t)BN * QN * CN * 4);
  float*  scd     = (float*)alloc((size_t)BN * QN * CN * 4);
  float*  fpro_raw= (float*)alloc((size_t)BN * QN * CN * 4);
  float*  meanx   = (float*)alloc((size_t)BN * CN * 4);
  float*  mu      = (float*)alloc((size_t)BN * CN * 4);
  float*  rstd    = (float*)alloc((size_t)BN * CN * 4);
  float*  sc1     = (float*)alloc((size_t)BN * QN * PN * 4);
  float*  sc2     = (float*)alloc((size_t)BN * QN * PN * 4);
  float*  sc3     = (float*)alloc((size_t)BN * QN * N3 * 4);
  float*  s1cam   = (float*)alloc((size_t)BN * QN * 4);
  float*  s2cam   = (float*)alloc((size_t)BN * QN * 4);
  float*  s3cam   = (float*)alloc((size_t)BN * NI * QN * 4);
  float*  sumt    = (float*)alloc((size_t)BN * QN * 4);
  float*  tmx     = (float*)alloc((size_t)BN * QN * 4);
  float*  S1v     = (float*)alloc((size_t)BN * QN * 4);
  float*  S2v     = (float*)alloc((size_t)BN * QN * 4);
  float*  wsm     = (float*)alloc((size_t)QN * 4);
  int*    b3      = (int*)alloc((size_t)BN * NI * 4);

  const long QC = (long)QN * CN;

  // ---- prep ----
  k_prep<<<BN * PN, 256, 0, stream>>>(x, xs_bf, rn);
  k_cam<<<BN, 256, 0, stream>>>(cam_ids, cam_tab, cam);
  k_perm<<<1, 32, 0, stream>>>(sub, b3);
  k_wsm<<<1, 32, 0, stream>>>(weights, wsm);

  // ---- catcher cosine self-attention over tokens (shared by all catchers) ----
  k_gram<<<dim3((PN / 16) * (PN / 16), BN), 32, 0, stream>>>(xs_bf, rn, S);
  k_softmax<<<BN * PN, 256, 0, stream>>>(S, PN);
  k_g2<<<dim3((PN / 16) * (CN / 16), BN), 32, 0, stream>>>(S, xs_bf, g2_bf);

  // ---- combined projection matrices: (Wq^T @ Wg), stored transposed ----
  k_wcomb_lds<<<dim3(CN / 64, CN / 64), 256, 0, stream>>>(Wcq, Wcg, WcombT);
  k_wcomb_lds<<<dim3(CN / 64, CN / 64), 256, 0, stream>>>(Wdq, Wdg, WdcombT);

  // ---- catcher 1: queries = query_v (shared across batch) ----
  k_rowgemm<<<dim3(CN / 16, 1), 32, 0, stream>>>(query_v, 0, WcombT, u1);
  k_camdot<<<BN * QN, 256, 0, stream>>>(u1, 0, cam, nullptr, 1, s1cam);
  k_scores16<<<dim3(PN / 16, BN), 32, 0, stream>>>(u1, 0, g2_bf, nullptr, 1, s1cam, sc1);
  k_softmax<<<BN * QN, 256, 0, stream>>>(sc1, PN);
  k_attout<<<dim3(CN / 16, BN), 32, 0, stream>>>(sc1, g2_bf, nullptr, 1, f_rel);

  // ---- deltaor (gd never materialized; LN folded into epilogues) ----
  k_ln<<<BN * QN, 256, 0, stream>>>(f_rel, qln);
  k_meanx<<<dim3(CN / 256, BN), 256, 0, stream>>>(f_rel, meanx);
  k_rowgemm<<<dim3(CN / 16, BN), 32, 0, stream>>>(qln, QC, WdcombT, tmat);
  k_dstats<<<dim3(CN, BN), 256, 0, stream>>>(proto, meanx, mu, rstd);
  k_taux<<<BN * QN, 256, 0, stream>>>(tmat, meanx, sumt, tmx);
  k_dscores<<<dim3(CN / 16, BN), 32, 0, stream>>>(tmat, proto, mu, rstd, sumt, tmx, scd);
  k_softmax<<<BN * QN, 256, 0, stream>>>(scd, CN);
  k_wmul<<<BN * QN, 256, 0, stream>>>(scd, rstd, mu, S1v, S2v);
  k_outd<<<dim3(CN / 16, BN), 32, 0, stream>>>(scd, proto, f_rel, meanx, S1v, S2v, wsm,
                                               fpro_raw, f_pro);

  // ---- catcher 2 (f_rec): same tokens, queries = f_pro_raw ----
  k_rowgemm<<<dim3(CN / 16, BN), 32, 0, stream>>>(fpro_raw, QC, WcombT, u2);
  k_camdot<<<BN * QN, 256, 0, stream>>>(u2, QC, cam, nullptr, 1, s2cam);
  k_scores16<<<dim3(PN / 16, BN), 32, 0, stream>>>(u2, QC, g2_bf, nullptr, 1, s2cam, sc2);
  k_softmax<<<BN * QN, 256, 0, stream>>>(sc2, PN);
  k_attout<<<dim3(CN / 16, BN), 32, 0, stream>>>(sc2, g2_bf, nullptr, 1, f_rec);

  // ---- catcher 3 (f_cor): gathered cross-instance token groups via b3 ----
  k_camdot<<<BN * NI * QN, 256, 0, stream>>>(u2, QC, cam, b3, NI, s3cam);
  k_scores16<<<dim3(NI * (PN / 16), BN), 32, 0, stream>>>(u2, QC, g2_bf, b3, NI, s3cam, sc3);
  k_softmax<<<BN * QN, 256, 0, stream>>>(sc3, N3);
  k_attout<<<dim3(CN / 16, BN), 32, 0, stream>>>(sc3, g2_bf, b3, NI, f_cor);
}
